// truncated_krylov_layer_16724602650838
// MI455X (gfx1250) — compile-verified, matrix operand order checked by reading, not run on hardware
//
#include <hip/hip_runtime.h>

typedef __attribute__((ext_vector_type(16))) __bf16 v16bf;
typedef __attribute__((ext_vector_type(8)))  float  v8f;

// ---------------------------------------------------------------------------
// Kernel 1: initialize Krylov matrix Kmat[N][1024]: slice 0 = input, rest = 0.
// One block per row, each thread writes one float4 (256 float4 per 1024-f row).
// ---------------------------------------------------------------------------
__global__ __launch_bounds__(256) void init_kmat(const float* __restrict__ X,
                                                 float* __restrict__ Kmat) {
  const int row = blockIdx.x;
  const int t   = threadIdx.x;              // float4 index 0..255
  float4 v;
  if (t < 32) {                             // first 128 floats = input row
    v = ((const float4*)(X + (size_t)row * 128))[t];
  } else {
    v = make_float4(0.f, 0.f, 0.f, 0.f);
  }
  ((float4*)(Kmat + (size_t)row * 1024))[t] = v;
}

// ---------------------------------------------------------------------------
// Kernel 2: SpMM step  Kmat[d][colOut..] += w * Kmat[s][colIn..]
// One wave (32 lanes) per edge; each lane handles 4 consecutive features.
// ---------------------------------------------------------------------------
__global__ __launch_bounds__(256) void spmm_step(const int*   __restrict__ esrc,
                                                 const int*   __restrict__ edst,
                                                 const float* __restrict__ ew,
                                                 float*       __restrict__ Kmat,
                                                 int colIn, int colOut, int E) {
  const int gid  = blockIdx.x * 256 + threadIdx.x;
  const int e    = gid >> 5;
  const int lane = gid & 31;
  if (e >= E) return;
  const int   s = esrc[e];
  const int   d = edst[e];
  const float w = ew[e];
  const float4 x = *(const float4*)(Kmat + (size_t)s * 1024 + colIn + lane * 4);
  float* o = Kmat + (size_t)d * 1024 + colOut + lane * 4;
  __hip_atomic_fetch_add(o + 0, w * x.x, __ATOMIC_RELAXED, __HIP_MEMORY_SCOPE_AGENT);
  __hip_atomic_fetch_add(o + 1, w * x.y, __ATOMIC_RELAXED, __HIP_MEMORY_SCOPE_AGENT);
  __hip_atomic_fetch_add(o + 2, w * x.z, __ATOMIC_RELAXED, __HIP_MEMORY_SCOPE_AGENT);
  __hip_atomic_fetch_add(o + 3, w * x.w, __ATOMIC_RELAXED, __HIP_MEMORY_SCOPE_AGENT);
}

// ---------------------------------------------------------------------------
// Kernel 3: pack W [1024 x 128] into per-lane WMMA B-fragment layout (bf16
// hi + residual lo). Fragment index: ((chunk*8 + ntile)*32 + lane)*16 + 2v+j.
// Lane n (0..15) holds column n of the 32x16 B chunk; lanes 16..31 hold the
// same columns but the K+8 / K+24 groups (mirror of the 16-bit A layout).
// ---------------------------------------------------------------------------
__global__ __launch_bounds__(256) void build_wfrag(const float* __restrict__ W,
                                                   __bf16* __restrict__ Bhi,
                                                   __bf16* __restrict__ Blo) {
  const int gid = blockIdx.x * 256 + threadIdx.x;   // 0 .. 65535
  if (gid >= 32 * 8 * 32 * 8) return;
  const int v    = gid & 7;
  const int lane = (gid >> 3) & 31;
  const int t    = (gid >> 8) & 7;
  const int c    = gid >> 11;
  const int col  = t * 16 + (lane & 15);
  const int sel  = lane >> 4;
  const int k0   = c * 32 + sel * 8 + ((v >= 4) ? 16 : 0) + (v & 3) * 2;
  const float w0 = W[(size_t)k0 * 128 + col];
  const float w1 = W[(size_t)(k0 + 1) * 128 + col];
  const __bf16 h0 = (__bf16)w0, h1 = (__bf16)w1;
  const size_t base = (size_t)gid * 2;
  Bhi[base + 0] = h0;
  Bhi[base + 1] = h1;
  Blo[base + 0] = (__bf16)(w0 - (float)h0);
  Blo[base + 1] = (__bf16)(w1 - (float)h1);
}

// ---------------------------------------------------------------------------
// Kernel 4: GEMM  out[N x 128] = Kmat[N x 1024] @ W + bias  via split-bf16
// WMMA (AhBh + AlBh + AhBl, f32 accumulate). Block = 8 waves = one 16-row
// M-tile x all 128 columns; wave w owns output columns [16w, 16w+16).
// ---------------------------------------------------------------------------
__global__ __launch_bounds__(256) void krylov_gemm(const float*  __restrict__ Kmat,
                                                   const __bf16* __restrict__ Bhi,
                                                   const __bf16* __restrict__ Blo,
                                                   const float*  __restrict__ bias,
                                                   float*        __restrict__ out) {
  __shared__ float2 As2[16 * 512];            // 16 rows x 1024 f32, pair units (64 KB)
  const int tid   = threadIdx.x;
  const int mbase = blockIdx.x * 16;

  // Stage A tile into LDS with XOR pair-swizzle (pair ^ row) to avoid 16-way
  // bank conflicts on the 4 KB row stride.
  const float4* gsrc = (const float4*)(Kmat + (size_t)mbase * 1024);
  for (int i = tid; i < 16 * 256; i += 256) {
    const int r  = i >> 8;
    const int c4 = i & 255;
    const float4 v = gsrc[r * 256 + c4];
    float2* base = As2 + r * 512;
    base[(c4 * 2 + 0) ^ r] = make_float2(v.x, v.y);
    base[(c4 * 2 + 1) ^ r] = make_float2(v.z, v.w);
  }
  __syncthreads();

  const int wave = tid >> 5;
  const int lane = tid & 31;
  const int row  = lane & 15;
  const int sel  = lane >> 4;

  v8f acc0 = {}, acc1 = {}, acc2 = {};

  for (int c = 0; c < 32; ++c) {
    // B fragments: one contiguous 32B load per lane from the packed layout.
    const size_t fidx = ((size_t)((c * 8 + wave) * 32 + lane)) << 4;
    const v16bf bh = *(const v16bf*)(Bhi + fidx);
    const v16bf bl = *(const v16bf*)(Blo + fidx);

    // A fragment: gather 8 f32 pairs from LDS, convert to bf16 hi + lo.
    v16bf ah, al;
    const int pairBase = c * 16 + sel * 4;    // (c*32 + sel*8) / 2
    const float2* arow = As2 + row * 512;
#pragma unroll
    for (int v = 0; v < 8; ++v) {
      const int poff = ((v >= 4) ? 8 : 0) + (v & 3);
      const float2 x = arow[(pairBase + poff) ^ row];
      const __bf16 h0 = (__bf16)x.x, h1 = (__bf16)x.y;
      ah[2 * v + 0] = h0;
      ah[2 * v + 1] = h1;
      al[2 * v + 0] = (__bf16)(x.x - (float)h0);
      al[2 * v + 1] = (__bf16)(x.y - (float)h1);
    }

    // Three independent accumulation chains (reduces WMMA RAW hazard stalls).
    acc0 = __builtin_amdgcn_wmma_f32_16x16x32_bf16(false, ah, false, bh,
                                                   (short)0, acc0, false, false);
    acc1 = __builtin_amdgcn_wmma_f32_16x16x32_bf16(false, al, false, bh,
                                                   (short)0, acc1, false, false);
    acc2 = __builtin_amdgcn_wmma_f32_16x16x32_bf16(false, ah, false, bl,
                                                   (short)0, acc2, false, false);
  }

  const int col = wave * 16 + row;
  const float b = bias[col];
#pragma unroll
  for (int j = 0; j < 8; ++j) {
    const int r = mbase + j + sel * 8;        // C layout: VGPR j -> M=j / M=j+8
    out[(size_t)r * 128 + col] = acc0[j] + acc1[j] + acc2[j] + b;
  }
}

// ---------------------------------------------------------------------------
// Host-side launcher (graph-capture safe: only kernel launches on `stream`).
// ---------------------------------------------------------------------------
extern "C" void kernel_launch(void* const* d_in, const int* in_sizes, int n_in,
                              void* d_out, int out_size, void* d_ws, size_t ws_size,
                              hipStream_t stream) {
  const float* input = (const float*)d_in[0];   // [N, 128]
  const int*   esrc  = (const int*)d_in[1];     // [E]
  const int*   edst  = (const int*)d_in[2];     // [E]
  const float* ew    = (const float*)d_in[3];   // [E]
  const float* W     = (const float*)d_in[4];   // [1024, 128]
  const float* bias  = (const float*)d_in[5];   // [128]
  float*       out   = (float*)d_out;           // [N, 128]

  const int N = in_sizes[0] / 128;              // 50000
  const int E = in_sizes[1];                    // 800000

  // Workspace layout: Kmat [N x 1024] f32, then packed W fragments (bf16).
  char* ws = (char*)d_ws;
  float* Kmat = (float*)ws;
  const size_t kmatBytes = (size_t)N * 1024 * sizeof(float);
  __bf16* Bhi = (__bf16*)(ws + kmatBytes);
  __bf16* Blo = Bhi + (size_t)32 * 8 * 32 * 16; // 131072 bf16 each

  // 1) Kmat slice 0 = input, slices 1..7 = 0 (re-done every call; d_ws is
  //    not re-poisoned between replays, so we always rebuild from scratch).
  init_kmat<<<N, 256, 0, stream>>>(input, Kmat);

  // 2) Pack W into WMMA fragment layout (hi/lo bf16 split).
  build_wfrag<<<(32 * 8 * 32 * 8 + 255) / 256, 256, 0, stream>>>(W, Bhi, Blo);

  // 3) Seven SpMM steps: slice i -> slice i+1.
  const int spmmThreads = E * 32;
  const int spmmBlocks  = (spmmThreads + 255) / 256;
  for (int i = 0; i < 7; ++i) {
    spmm_step<<<spmmBlocks, 256, 0, stream>>>(esrc, edst, ew, Kmat,
                                              i * 128, (i + 1) * 128, E);
  }

  // 4) Dense GEMM with WMMA: out = Kmat @ W + bias.
  krylov_gemm<<<N / 16, 256, 0, stream>>>(Kmat, Bhi, Blo, bias, out);
}
